// OODGuard_65377992180537
// MI455X (gfx1250) — matrix operand order repaired and checked by reading, not compiled, not run on hardware
//
#include <hip/hip_runtime.h>
#include <hip/hip_bf16.h>

// ---------------------------------------------------------------------------
// CDNA5 (gfx1250) fused kNN-OOD guard.
//  - bf16 WMMA 16x16x32 for both Gram matrices (S.S^T and Z.S^T)
//  - A tile staged by the Tensor Data Mover (tensor_load_to_lds, TENSORcnt)
//  - B tiles streamed with GLOBAL_LOAD_ASYNC_TO_LDS_B128, double-buffered
//    K-chunks overlapped with WMMA (ASYNCcnt)
//  - distance tiles folded into per-row top-k lists in registers
//  - exact 0.99 quantile via bit-space binary search
// ---------------------------------------------------------------------------

typedef __attribute__((ext_vector_type(16))) __bf16   v16bf;
typedef __attribute__((ext_vector_type(8)))  float    v8f;
typedef __attribute__((ext_vector_type(4)))  unsigned v4u;
typedef __attribute__((ext_vector_type(8)))  unsigned v8u;

union Frag { v16bf bf; uint4 q[2]; };

static __device__ __forceinline__ unsigned short f2bf(float f) {
  unsigned u = __builtin_bit_cast(unsigned, f);
  unsigned r = u + 0x7FFFu + ((u >> 16) & 1u);   // round-to-nearest-even
  return (unsigned short)(r >> 16);
}

static __device__ __forceinline__ unsigned rfl(unsigned x) {
  return (unsigned)__builtin_amdgcn_readfirstlane((int)x);
}

static __device__ __forceinline__ unsigned lds_addr_of(const void* p) {
  return (unsigned)(unsigned long long)p;   // low 32 bits = LDS byte offset
}

// --- TDM: load `rows` x 512 bf16 (row-major, stride 512) into LDS ----------
static __device__ __forceinline__ void tdm_load_rows512(const unsigned short* g,
                                                        unsigned lds_byte,
                                                        unsigned rows) {
  const unsigned long long ga = (unsigned long long)g;
  v4u g0;
  g0[0] = rfl(1u);                                        // count=1, no gather
  g0[1] = rfl(lds_byte);                                  // lds_addr
  g0[2] = rfl((unsigned)(ga & 0xFFFFFFFFull));            // global_addr lo
  g0[3] = rfl((unsigned)((ga >> 32) & 0x1FFFFFFu) | (2u << 30)); // hi | type=2
  v8u g1;
  g1[0] = rfl(1u << 16);          // data_size = 2 bytes
  g1[1] = rfl(512u << 16);        // tensor_dim0 = 512 (lo16 @ bits 63:48)
  g1[2] = rfl(rows << 16);        // dim0 hi = 0 | tensor_dim1 lo16 = rows
  g1[3] = rfl(512u << 16);        // dim1 hi = 0 | tile_dim0 = 512
  g1[4] = rfl(rows);              // tile_dim1 = rows | tile_dim2 = 0
  g1[5] = rfl(512u);              // tensor_dim0_stride lo32 = 512
  g1[6] = rfl(0u);                // stride hi | tensor_dim1_stride lo
  g1[7] = rfl(0u);                // tensor_dim1_stride hi
  asm volatile("tensor_load_to_lds %0, %1" :: "s"(g0), "s"(g1) : "memory");
  __builtin_amdgcn_s_wait_tensorcnt(0);
}

// --- async global->LDS stage of one 16x64 bf16 B chunk (2KB, 4 b128/lane) --
static __device__ __forceinline__ void stage_async(const unsigned short* gbase,
                                                   unsigned lds_base, int lane) {
#pragma unroll
  for (int j = 0; j < 4; ++j) {
    const int idx = j * 32 + lane;                 // 0..127 (16 rows x 8 xb128)
    const void* gp = gbase + (size_t)(idx >> 3) * 512 + (size_t)(idx & 7) * 8;
    const unsigned lp = lds_base + (unsigned)idx * 16u;
    asm volatile("global_load_async_to_lds_b128 %0, %1, off"
                 :: "v"(lp), "v"(gp) : "memory");
  }
}

// ---------------------------------------------------------------------------
// Kernel 1: row-normalize f32 [rows x 512] -> bf16, and emit ||row_norm||^2.
// ---------------------------------------------------------------------------
__global__ void __launch_bounds__(128)
normalize_bf16_k(const float* __restrict__ src, unsigned short* __restrict__ dst,
                 float* __restrict__ sq) {
  const int row = blockIdx.x;
  const float4 v = ((const float4*)(src + (size_t)row * 512))[threadIdx.x];
  float p = v.x * v.x + v.y * v.y + v.z * v.z + v.w * v.w;

  __shared__ float red[128];
  red[threadIdx.x] = p;
  __syncthreads();
  for (int s = 64; s > 0; s >>= 1) {
    if (threadIdx.x < s) red[threadIdx.x] += red[threadIdx.x + s];
    __syncthreads();
  }
  const float ss  = red[0];
  const float inv = 1.0f / (sqrtf(ss) + 1e-8f);

  unsigned lo = (unsigned)f2bf(v.x * inv) | ((unsigned)f2bf(v.y * inv) << 16);
  unsigned hi = (unsigned)f2bf(v.z * inv) | ((unsigned)f2bf(v.w * inv) << 16);
  uint2 pk; pk.x = lo; pk.y = hi;
  ((uint2*)(dst + (size_t)row * 512))[threadIdx.x] = pk;
  if (threadIdx.x == 0) sq[row] = ss * inv * inv;
}

// ---------------------------------------------------------------------------
// Kernel 2: fused GEMM + streaming top-k.
// ---------------------------------------------------------------------------
template <int KL, bool SELF>
__global__ void __launch_bounds__(256)
knn_k(const unsigned short* __restrict__ Abf, const float* __restrict__ sqA,
      const unsigned short* __restrict__ Sbf, const float* __restrict__ sqS,
      float* __restrict__ outAvg, int ncols) {
  // LDS: A tile 16x512 bf16 (16KB) | 8 waves x double-buffered 16x64 B chunk
  //      (2 x 2KB each = 32KB) | overlay: gram tiles (8KB) / merge (11KB)
  __shared__ __align__(16) unsigned char smem[16384 + 32768 + 11264];
  unsigned short* As = (unsigned short*)smem;
  unsigned short* Bs = (unsigned short*)(smem + 16384);
  float*          Ds = (float*)(smem + 16384 + 32768);
  float*          Mg = Ds;  // merge buffer reuses gram-tile region (after sync)

  const int tid   = threadIdx.x;
  const int lane  = tid & 31;
  const int wave  = tid >> 5;
  const int mbase = blockIdx.x * 16;

  // ---- stage A tile via TDM: each wave DMAs its 2 rows (2x512 bf16) ----
  tdm_load_rows512(Abf + (size_t)(mbase + 2 * wave) * 512,
                   lds_addr_of(As) + (unsigned)wave * 2048u, 2u);
  __syncthreads();

  const int   r    = lane & 15;   // row within tile (A/D), col within tile (B)
  const int   hi   = lane >> 4;
  const float rsq  = sqA[mbase + r];
  const int   rowg = mbase + r;

  // ---- preload all 16 K-step A fragments (loop-invariant) ----
  Frag fa[16];
  {
    const uint4* Aq = (const uint4*)As;
#pragma unroll
    for (int s = 0; s < 16; ++s) {
      fa[s].q[0] = Aq[(r * 512 + s * 32 + hi * 8) >> 3];
      fa[s].q[1] = Aq[(r * 512 + s * 32 + 16 + hi * 8) >> 3];
    }
  }

  float lst[KL];
#pragma unroll
  for (int i = 0; i < KL; ++i) lst[i] = 3.0e38f;

  unsigned short* Bw      = Bs + wave * 2048;                 // 4KB (2 bufs)
  const unsigned  bw_base = lds_addr_of(Bs) + (unsigned)wave * 4096u;
  float*          Dw      = Ds + wave * 256;                  // 16x16 f32

  const int ntiles = ncols >> 4;
  for (int t = wave; t < ntiles; t += 8) {
    const int cbase = t * 16;
    const unsigned short* gB = Sbf + (size_t)cbase * 512;
    v8f acc = {};

    stage_async(gB, bw_base, lane);                 // chunk 0 in flight
#pragma unroll
    for (int kb = 0; kb < 8; ++kb) {                // K staged in 8x64 chunks
      if (kb < 7) {
        stage_async(gB + (kb + 1) * 64, bw_base + (((kb + 1) & 1) << 11), lane);
        asm volatile("s_wait_asynccnt 0x4" ::: "memory");  // chunk kb landed
      } else {
        asm volatile("s_wait_asynccnt 0x0" ::: "memory");
      }
      const uint4* Bq = (const uint4*)(Bw + ((kb & 1) << 10));
#pragma unroll
      for (int s2 = 0; s2 < 2; ++s2) {              // 2 x K=32 WMMA steps
        Frag fb;
        // B 32x16 bf16 layout: lanes<16 hold K 0..7,16..23, lanes>=16 hold K 8..15,24..31
        fb.q[0] = Bq[(r * 64 + s2 * 32 + hi * 8) >> 3];
        fb.q[1] = Bq[(r * 64 + s2 * 32 + 16 + hi * 8) >> 3];
        acc = __builtin_amdgcn_wmma_f32_16x16x32_bf16(
            false, fa[kb * 2 + s2].bf, false, fb.bf, (short)0, acc, false, false);
      }
    }

    // ---- bounce tile through LDS to remap lane->row, update top-k ----
#pragma unroll
    for (int p = 0; p < 8; ++p) Dw[(p + hi * 8) * 16 + r] = acc[p];
    __builtin_amdgcn_wave_barrier();
#pragma unroll
    for (int j = 0; j < 8; ++j) {
      const int   c    = hi * 8 + j;
      const float g    = Dw[r * 16 + c];
      const int   colg = cbase + c;
      const float d2   = rsq + sqS[colg] - 2.0f * g;
      float       d    = sqrtf(fmaxf(d2, 1e-12f));
      if (SELF && colg == rowg) d = 0.0f;       // self-distance exactly smallest
      if (d < lst[KL - 1]) {                    // branchless sorted insert
#pragma unroll
        for (int i = KL - 1; i >= 1; --i)
          lst[i] = fminf(fmaxf(d, lst[i - 1]), lst[i]);
        lst[0] = fminf(lst[0], d);
      }
    }
    __builtin_amdgcn_wave_barrier();
  }

  // ---- merge 16 partial lists per row (8 waves x 2 column-halves) ----
  __syncthreads();  // all waves done with gram-tile region
  {
    const int slot = wave * 2 + hi;
    float* dl = Mg + (slot * 16 + r) * KL;
#pragma unroll
    for (int i = 0; i < KL; ++i) dl[i] = lst[i];
  }
  __syncthreads();
  if (tid < 16) {
    float fl[KL];
#pragma unroll
    for (int i = 0; i < KL; ++i) fl[i] = 3.0e38f;
    for (int s = 0; s < 16; ++s) {
      const float* sl = Mg + (s * 16 + tid) * KL;
#pragma unroll
      for (int k = 0; k < KL; ++k) {
        const float d = sl[k];
        if (d < fl[KL - 1]) {
#pragma unroll
          for (int i = KL - 1; i >= 1; --i)
            fl[i] = fminf(fmaxf(d, fl[i - 1]), fl[i]);
          fl[0] = fminf(fl[0], d);
        }
      }
    }
    float sum = 0.0f;
#pragma unroll
    for (int i = (SELF ? 1 : 0); i < KL; ++i) sum += fl[i];  // drop self slot
    outAvg[mbase + tid] = sum / (float)(KL - (SELF ? 1 : 0));
  }
}

// ---------------------------------------------------------------------------
// Kernel 3: exact 0.99 quantile via bit-space binary search -> threshold.
// ---------------------------------------------------------------------------
__global__ void __launch_bounds__(1024)
quantile_k(const float* __restrict__ avg, float* __restrict__ dout, int n) {
  __shared__ float    v[12288];
  __shared__ unsigned cnt;
  for (int i = threadIdx.x; i < n; i += blockDim.x) v[i] = avg[i];
  __syncthreads();

  const float pos = 0.99f * (float)(n - 1);
  const int   k0  = (int)pos;
  const float fr  = pos - (float)k0;

  float kv[2];
  for (int which = 0; which < 2; ++which) {
    const unsigned need = (unsigned)(k0 + which + 1);
    unsigned lo = 0u, hiB = 0x7F800000u;     // [0, +inf)
    while (lo < hiB) {
      const unsigned mid = (lo + hiB) >> 1;
      const float    mv  = __builtin_bit_cast(float, mid);
      if (threadIdx.x == 0) cnt = 0u;
      __syncthreads();
      unsigned c = 0;
      for (int i = threadIdx.x; i < n; i += blockDim.x) c += (v[i] <= mv);
      atomicAdd(&cnt, c);
      __syncthreads();
      if (cnt >= need) hiB = mid; else lo = mid + 1;
      __syncthreads();
    }
    kv[which] = __builtin_bit_cast(float, lo);
  }
  if (threadIdx.x == 0) {
    const float base = kv[0] + fr * (kv[1] - kv[0]);
    dout[0] = base * 1.5f;  // SENSITIVITY
  }
}

// ---------------------------------------------------------------------------
// Kernel 4: emit avg_q and over = avg_q > threshold (reads dout[0]).
// ---------------------------------------------------------------------------
__global__ void finalize_k(const float* __restrict__ avgQ, float* __restrict__ dout) {
  const int i = blockIdx.x * blockDim.x + threadIdx.x;
  if (i < 1024) {
    const float thr = dout[0];
    const float a   = avgQ[i];
    dout[1 + i]    = a;
    dout[1025 + i] = (a > thr) ? 1.0f : 0.0f;
  }
}

// ---------------------------------------------------------------------------
// Kernel 5: column-wise min/max of global_embedding vs bounds.
// ---------------------------------------------------------------------------
__global__ void __launch_bounds__(256)
bounds_k(const float* __restrict__ ge, const float* __restrict__ gmin,
         const float* __restrict__ gmax, float* __restrict__ dout) {
  const int d = threadIdx.x;  // 256 columns, coalesced across lanes
  float mn = 3.0e38f, mx = -3.0e38f;
  for (int r = 0; r < 1024; ++r) {
    const float v = ge[(size_t)r * 256 + d];
    mn = fminf(mn, v);
    mx = fmaxf(mx, v);
  }
  dout[2049 + d] = (mn < gmin[d]) ? 1.0f : 0.0f;
  dout[2305 + d] = (mx > gmax[d]) ? 1.0f : 0.0f;
}

// ---------------------------------------------------------------------------
extern "C" void kernel_launch(void* const* d_in, const int* in_sizes, int n_in,
                              void* d_out, int out_size, void* d_ws, size_t ws_size,
                              hipStream_t stream) {
  (void)in_sizes; (void)n_in; (void)out_size; (void)ws_size;
  const float* ge   = (const float*)d_in[0];  // [1024, 256]
  const float* gl   = (const float*)d_in[1];  // [1024, 512]
  const float* geo  = (const float*)d_in[2];  // [12288, 512]
  const float* gmin = (const float*)d_in[3];  // [256]
  const float* gmax = (const float*)d_in[4];  // [256]
  float* out = (float*)d_out;                 // [1 + 1024 + 1024 + 256 + 256]

  char* ws = (char*)d_ws;
  unsigned short* Sbf = (unsigned short*)ws;                        // 12.6 MB
  unsigned short* Zbf = (unsigned short*)(ws + 12582912);           // 1 MB
  float* sqS  = (float*)(ws + 12582912 + 1048576);                  // 12288
  float* sqZ  = sqS + 12288;                                        // 1024
  float* avgK = sqZ + 1024;                                         // 12288
  float* avgQ = avgK + 12288;                                       // 1024

  normalize_bf16_k<<<12288, 128, 0, stream>>>(geo, Sbf, sqS);
  normalize_bf16_k<<<1024, 128, 0, stream>>>(gl, Zbf, sqZ);
  knn_k<11, true ><<<768, 256, 0, stream>>>(Sbf, sqS, Sbf, sqS, avgK, 12288);
  knn_k<10, false><<<64,  256, 0, stream>>>(Zbf, sqZ, Sbf, sqS, avgQ, 12288);
  quantile_k<<<1, 1024, 0, stream>>>(avgK, out, 12288);
  finalize_k<<<4, 256, 0, stream>>>(avgQ, out);
  bounds_k<<<1, 256, 0, stream>>>(ge, gmin, gmax, out);
}
